// CosVQ_1657857376703
// MI455X (gfx1250) — matrix operand order repaired and checked by reading, not compiled, not run on hardware
//
#include <hip/hip_runtime.h>
#include <hip/hip_bf16.h>

// ---------------- problem constants ----------------
#define NTOK   16384          // 4*4096 rows of z
#define DIM    128            // embedding dim (K of the GEMM)
#define NCODE  8192           // codebook size
#define INV_T  10.0f          // 1/temperature ; |score|<=1 so exp(10*s) never overflows fp32
#define BETA   0.25f
#define EPSN   1e-12f
#define CSPLIT 4              // pass1 column splits  -> 512 blocks
#define RSPLIT 8              // pass2 row slices     -> 512 blocks

typedef _Float16 v16h __attribute__((ext_vector_type(16)));
typedef _Float16 h8   __attribute__((ext_vector_type(8)));
typedef _Float16 h4   __attribute__((ext_vector_type(4)));
typedef float    v8f  __attribute__((ext_vector_type(8)));

static __device__ __forceinline__ v16h frag_from(const _Float16* p_lo, const _Float16* p_hi) {
    h8 lo = *(const h8*)p_lo;
    h8 hi = *(const h8*)p_hi;
    return __builtin_shufflevector(lo, hi, 0,1,2,3,4,5,6,7,8,9,10,11,12,13,14,15);
}

// ---------------------------------------------------------------------------
// Kernel 1: row L2-normalize fp32 -> f16   (one wave per row, 4 floats/lane)
// ---------------------------------------------------------------------------
__global__ void k_norm_rows(const float* __restrict__ src, _Float16* __restrict__ dst) {
    const int lane = threadIdx.x & 31;
    const int wave = threadIdx.x >> 5;
    const int row  = blockIdx.x * 8 + wave;

    const float4 x = ((const float4*)src)[row * 32 + lane];
    float ss = x.x * x.x + x.y * x.y + x.z * x.z + x.w * x.w;
    #pragma unroll
    for (int off = 16; off > 0; off >>= 1) ss += __shfl_xor(ss, off);
    const float inv = 1.0f / fmaxf(sqrtf(ss), EPSN);

    h4 o;
    o.x = (_Float16)(x.x * inv); o.y = (_Float16)(x.y * inv);
    o.z = (_Float16)(x.z * inv); o.w = (_Float16)(x.w * inv);
    ((h4*)dst)[row * 32 + lane] = o;
}

// ---------------------------------------------------------------------------
// Kernel 2: pass 1 GEMM — row-stationary, branch-free.
// Grid 512 = 128 row-blocks x 4 column splits.  Each wave owns 16 rows
// (A frags in regs), sweeps its 2048-column range in 16-col LDS tiles with a
// one-deep software pipeline.  Per lane: sumexp += exp(10*s) and cndmask
// argmax — no divergence.  Emits per-(split,row) partial sumexp/argmax.
// ---------------------------------------------------------------------------
__global__ void k_pass1(const _Float16* __restrict__ zn, const _Float16* __restrict__ en,
                        float* __restrict__ lsumP, float* __restrict__ bestSP,
                        int* __restrict__ bestIP) {
    __shared__ __align__(32) _Float16 bT[16 * DIM];   // [n][k] f16, 4KB

    const int tid    = threadIdx.x;
    const int lane   = tid & 31;
    const int wave   = tid >> 5;
    const int laneHi = (lane >> 4) & 1;
    const int lane15 = lane & 15;
    const int rowBlk = blockIdx.x >> 2;               // 0..127
    const int split  = blockIdx.x & (CSPLIT - 1);     // 0..3
    const int rowBase = rowBlk * 128 + wave * 16;
    const int t0   = split * (NCODE / 16 / CSPLIT);   // 128 tiles per split
    const int tEnd = t0 + (NCODE / 16 / CSPLIT);

    // ---- A fragments: 16x128 f16 for this wave's rows, 4 K-chunks in regs ----
    v16h aF[4];
    {
        const _Float16* arow = zn + (size_t)(rowBase + lane15) * DIM;
        #pragma unroll
        for (int kc = 0; kc < 4; ++kc) {
            const int kb = kc * 32;
            aF[kc] = frag_from(arow + kb + laneHi * 8, arow + kb + 16 + laneHi * 8);
        }
    }

    float lrun[8], bestS[8];
    int   bestI[8];
    #pragma unroll
    for (int v = 0; v < 8; ++v) { lrun[v] = 0.0f; bestS[v] = -INFINITY; bestI[v] = 0; }

    const int ldN = tid >> 4;          // 0..15 column within tile
    const int ldK = (tid & 15) * 8;    // 0..120 k-offset

    // software pipeline: stage tile t0, then loop storing/staging
    h8 stage = *(const h8*)&en[(size_t)(t0 * 16 + ldN) * DIM + ldK];

    for (int t = t0; t < tEnd; ++t) {
        __syncthreads();
        *(h8*)&bT[ldN * DIM + ldK] = stage;
        __syncthreads();

        const int tn = (t + 1 < tEnd) ? t + 1 : t;               // prefetch next tile
        stage = *(const h8*)&en[(size_t)(tn * 16 + ldN) * DIM + ldK];

        v8f c = {};
        #pragma unroll
        for (int kc = 0; kc < 4; ++kc) {
            const int kb = kc * 32;
            v16h bF = frag_from(&bT[lane15 * DIM + kb + laneHi * 8],
                                &bT[lane15 * DIM + kb + 16 + laneHi * 8]);
            c = __builtin_amdgcn_wmma_f32_16x16x32_f16(false, aF[kc], false, bF,
                                                       (short)0, c, false, false);
        }

        const int col = t * 16 + lane15;
        #pragma unroll
        for (int v = 0; v < 8; ++v) {
            const float s = c[v];
            lrun[v] += __expf(s * INV_T);                        // no max-shift needed
            const bool gt = s > bestS[v];                        // '>' keeps first max
            bestS[v] = gt ? s   : bestS[v];
            bestI[v] = gt ? col : bestI[v];
        }
    }

    // ---- merge the 16 column classes (width-16 butterflies, branch-free) ----
    #pragma unroll
    for (int off = 1; off < 16; off <<= 1) {
        #pragma unroll
        for (int v = 0; v < 8; ++v) {
            lrun[v] += __shfl_xor(lrun[v], off, 16);
            const float s2 = __shfl_xor(bestS[v], off, 16);
            const int   i2 = __shfl_xor(bestI[v], off, 16);
            const bool take = (s2 > bestS[v]) || (s2 == bestS[v] && i2 < bestI[v]);
            bestS[v] = take ? s2 : bestS[v];
            bestI[v] = take ? i2 : bestI[v];
        }
    }

    if (lane15 == 0) {
        #pragma unroll
        for (int v = 0; v < 8; ++v) {
            const int r = rowBase + v + 8 * laneHi;
            lsumP [split * NTOK + r] = lrun[v];
            bestSP[split * NTOK + r] = bestS[v];
            bestIP[split * NTOK + r] = bestI[v];
        }
    }
}

// ---------------------------------------------------------------------------
// Kernel 3: combine the CSPLIT column-range partials per row.
// ---------------------------------------------------------------------------
__global__ void k_combine(const float* __restrict__ lsumP, const float* __restrict__ bestSP,
                          const int* __restrict__ bestIP,
                          float* __restrict__ invL, int* __restrict__ idx) {
    const int r = blockIdx.x * 256 + threadIdx.x;
    float lsum = 0.0f, bs = -INFINITY;
    int bi = 0;
    #pragma unroll
    for (int s = 0; s < CSPLIT; ++s) {                 // ascending column order
        lsum += lsumP[s * NTOK + r];
        const float s2 = bestSP[s * NTOK + r];
        const int   i2 = bestIP[s * NTOK + r];
        const bool take = s2 > bs;                     // strict > keeps first max
        bs = take ? s2 : bs;
        bi = take ? i2 : bi;
    }
    invL[r] = 1.0f / lsum;
    idx[r]  = bi;
}

// ---------------------------------------------------------------------------
// Kernel 4: pass 2 GEMM — column-stationary, branch-free.
// Grid 512 = 64 column-blocks x 8 row slices.  Each wave owns 16 columns
// (B frags in regs); A strips + invL stream through LDS (pipelined).
// Each lane accumulates one column's Σ probs over its slice -> exclusive
// store into Pp[slice][col] (no atomics, deterministic).
// ---------------------------------------------------------------------------
__global__ void k_pass2(const _Float16* __restrict__ zn, const _Float16* __restrict__ en,
                        const float* __restrict__ invL, float* __restrict__ Pp) {
    __shared__ __align__(32) _Float16 aT[16 * DIM];
    __shared__ float invLS[16];

    const int tid    = threadIdx.x;
    const int lane   = tid & 31;
    const int wave   = tid >> 5;
    const int laneHi = (lane >> 4) & 1;
    const int lane15 = lane & 15;
    const int colBlk = blockIdx.x >> 3;               // 0..63
    const int slice  = blockIdx.x & (RSPLIT - 1);     // 0..7
    const int colBase = colBlk * 128 + wave * 16;
    const int myCol   = colBase + lane15;
    const int t0   = slice * (NTOK / 16 / RSPLIT);    // 128 row strips per slice
    const int tEnd = t0 + (NTOK / 16 / RSPLIT);

    // ---- B fragments for this wave's columns, 4 K-chunks in regs ----
    v16h bF[4];
    {
        const _Float16* brow = en + (size_t)myCol * DIM;
        #pragma unroll
        for (int kc = 0; kc < 4; ++kc) {
            const int kb = kc * 32;
            bF[kc] = frag_from(brow + kb + laneHi * 8, brow + kb + 16 + laneHi * 8);
        }
    }

    float colAcc = 0.0f;
    const int ldN = tid >> 4;
    const int ldK = (tid & 15) * 8;

    h8    stage  = *(const h8*)&zn[(size_t)(t0 * 16 + ldN) * DIM + ldK];
    float iStage = (tid < 16) ? invL[t0 * 16 + tid] : 0.0f;

    for (int t = t0; t < tEnd; ++t) {
        __syncthreads();
        *(h8*)&aT[ldN * DIM + ldK] = stage;
        if (tid < 16) invLS[tid] = iStage;
        __syncthreads();

        const int tn = (t + 1 < tEnd) ? t + 1 : t;
        stage  = *(const h8*)&zn[(size_t)(tn * 16 + ldN) * DIM + ldK];
        iStage = (tid < 16) ? invL[tn * 16 + tid] : 0.0f;

        v8f c = {};
        #pragma unroll
        for (int kc = 0; kc < 4; ++kc) {
            const int kb = kc * 32;
            v16h aF = frag_from(&aT[lane15 * DIM + kb + laneHi * 8],
                                &aT[lane15 * DIM + kb + 16 + laneHi * 8]);
            c = __builtin_amdgcn_wmma_f32_16x16x32_f16(false, aF, false, bF[kc],
                                                       (short)0, c, false, false);
        }

        #pragma unroll
        for (int v = 0; v < 8; ++v) {
            const int vloc = v + 8 * laneHi;          // local row 0..15
            colAcc += __expf(c[v] * INV_T) * invLS[vloc];   // prob, fused scale
        }
    }

    // lanes L and L+16 hold the same column: pair-combine, lanes 0..15 store.
    const float other = __shfl_xor(colAcc, 16);
    if (laneHi == 0) Pp[slice * NCODE + myCol] = colAcc + other;
}

// ---------------------------------------------------------------------------
// Kernel 5: gather z_q = W[idx], per-block commit partial, int histogram.
// ---------------------------------------------------------------------------
__global__ void k_gather(const float* __restrict__ z, const float* __restrict__ W,
                         const int* __restrict__ idx, float* __restrict__ out,
                         float* __restrict__ commitPart, int* __restrict__ counts) {
    __shared__ float part[8];
    const int lane = threadIdx.x & 31;
    const int wave = threadIdx.x >> 5;
    const int row  = blockIdx.x * 8 + wave;
    const int code = idx[row];

    const float4 zx = ((const float4*)z)[row * 32 + lane];
    const float4 wx = ((const float4*)W)[(size_t)code * 32 + lane];
    ((float4*)out)[row * 32 + lane] = wx;

    const float dx = wx.x - zx.x, dy = wx.y - zx.y, dz = wx.z - zx.z, dw = wx.w - zx.w;
    float s = dx * dx + dy * dy + dz * dz + dw * dw;
    #pragma unroll
    for (int off = 16; off > 0; off >>= 1) s += __shfl_xor(s, off);

    if (lane == 0) {
        part[wave] = s;
        atomicAdd(&counts[code], 1);                  // integer: order-independent
    }
    __syncthreads();
    if (threadIdx.x == 0) {
        float t = 0.0f;
        #pragma unroll
        for (int w = 0; w < 8; ++w) t += part[w];
        commitPart[blockIdx.x] = t;
    }
}

// ---------------------------------------------------------------------------
// Kernel 6: finalize the three scalars.
// ---------------------------------------------------------------------------
__global__ void k_finalize(const float* __restrict__ commitPart, const int* __restrict__ counts,
                           const float* __restrict__ Pp, float* __restrict__ out) {
    __shared__ float red[256];
    const int tid = threadIdx.x;

    auto block_sum = [&](float v) -> float {
        red[tid] = v; __syncthreads();
        for (int s = 128; s > 0; s >>= 1) { if (tid < s) red[tid] += red[tid + s]; __syncthreads(); }
        const float r = red[0]; __syncthreads();
        return r;
    };

    float cs = 0.0f;
    for (int i = tid; i < NTOK / 8; i += 256) cs += commitPart[i];

    float ps = 0.0f, es = 0.0f;
    const float invN = 1.0f / (float)NTOK;
    for (int k = tid; k < NCODE; k += 256) {
        const float em = (float)counts[k] * invN;
        ps += em * __logf(em + 1e-8f);                // 0*log(1e-8) == 0, matches jnp
        float pt = 0.0f;
        #pragma unroll
        for (int s = 0; s < RSPLIT; ++s) pt += Pp[s * NCODE + k];
        const float pa = pt * invN + 1e-8f;
        es += pa * __logf(pa);
    }

    const float commit = block_sum(cs);
    const float perp   = block_sum(ps);
    const float ent    = block_sum(es);

    if (tid == 0) {
        const size_t base = (size_t)NTOK * DIM;
        out[base + 0] = (1.0f + BETA) * commit / (float)((size_t)NTOK * DIM);
        out[base + 1] = __expf(-perp);
        out[base + 2] = -ent;
    }
}

// ---------------------------------------------------------------------------
extern "C" void kernel_launch(void* const* d_in, const int* in_sizes, int n_in,
                              void* d_out, int out_size, void* d_ws, size_t ws_size,
                              hipStream_t stream) {
    const float* z = (const float*)d_in[0];   // [4,4096,128]
    const float* W = (const float*)d_in[1];   // [8192,128]
    float* out = (float*)d_out;               // z_q (N*D) | commit | perplexity | entropy

    // ---- workspace carve-out ----
    char* p = (char*)d_ws;
    _Float16* zn   = (_Float16*)p; p += (size_t)NTOK  * DIM * sizeof(_Float16);   // 4 MB
    _Float16* en   = (_Float16*)p; p += (size_t)NCODE * DIM * sizeof(_Float16);   // 2 MB
    float* lsumP   = (float*)p;    p += (size_t)CSPLIT * NTOK * sizeof(float);
    float* bestSP  = (float*)p;    p += (size_t)CSPLIT * NTOK * sizeof(float);
    int*   bestIP  = (int*)p;      p += (size_t)CSPLIT * NTOK * sizeof(int);
    float* invL    = (float*)p;    p += (size_t)NTOK  * sizeof(float);
    int*   idx     = (int*)p;      p += (size_t)NTOK  * sizeof(int);
    float* Pp      = (float*)p;    p += (size_t)RSPLIT * NCODE * sizeof(float);
    int*   counts  = (int*)p;      p += (size_t)NCODE * sizeof(int);
    float* cPart   = (float*)p;    p += (size_t)(NTOK / 8) * sizeof(float);

    hipMemsetAsync(counts, 0, NCODE * sizeof(int), stream);   // only buffer that accumulates

    k_norm_rows<<<NTOK  / 8, 256, 0, stream>>>(z, zn);
    k_norm_rows<<<NCODE / 8, 256, 0, stream>>>(W, en);
    k_pass1    <<<(NTOK / 128) * CSPLIT, 256, 0, stream>>>(zn, en, lsumP, bestSP, bestIP);
    k_combine  <<<NTOK / 256, 256, 0, stream>>>(lsumP, bestSP, bestIP, invL, idx);
    k_pass2    <<<(NCODE / 128) * RSPLIT, 256, 0, stream>>>(zn, en, invL, Pp);
    k_gather   <<<NTOK / 8, 256, 0, stream>>>(z, W, idx, out, cPart, counts);
    k_finalize <<<1, 256, 0, stream>>>(cPart, counts, Pp, out);
}
